// MultiHeadSelfAttention_62861141344317
// MI455X (gfx1250) — matrix operand order repaired and checked by reading, not compiled, not run on hardware
//
#include <hip/hip_runtime.h>
#include <hip/hip_bf16.h>

// ---------- CDNA5 WMMA types ----------
typedef __attribute__((ext_vector_type(16))) __bf16       v16bf;
typedef __attribute__((ext_vector_type(8)))  float        v8f;
typedef __attribute__((ext_vector_type(8)))  unsigned int v8u;
typedef __attribute__((ext_vector_type(4)))  unsigned int v4u;
typedef __attribute__((ext_vector_type(4)))  int          i32x4;

typedef __attribute__((address_space(1))) i32x4 gvec4; // global int4
typedef __attribute__((address_space(3))) i32x4 lvec4; // LDS int4

// ---------- async global->LDS path (gfx1250), with safe fallback ----------
#if defined(__has_builtin)
#if __has_builtin(__builtin_amdgcn_global_load_async_to_lds_b128) && \
    __has_builtin(__builtin_amdgcn_s_wait_asynccnt)
#define USE_ASYNC_LDS 1
#endif
#endif
#ifndef USE_ASYNC_LDS
#define USE_ASYNC_LDS 0
#endif

__device__ __forceinline__ void copy16_g2l(const unsigned short* g, unsigned short* l) {
#if USE_ASYNC_LDS
  __builtin_amdgcn_global_load_async_to_lds_b128(
      (gvec4*)(uintptr_t)g, (lvec4*)(uintptr_t)l, 0, 0);
#else
  *(v4u*)l = *(const v4u*)g;   // sync fallback: global_load_b128 + ds_store_b128
#endif
}

template <int N> __device__ __forceinline__ void wait_async_le() {
#if USE_ASYNC_LDS
  __builtin_amdgcn_s_wait_asynccnt(N);
#endif
}

// ---------- scalar helpers ----------
__device__ __forceinline__ v16bf as_v16bf(v8u u) { return __builtin_bit_cast(v16bf, u); }

__device__ __forceinline__ unsigned short f32_to_bf16_bits(float f) {
  unsigned int u = __float_as_uint(f);
  u += 0x7FFFu + ((u >> 16) & 1u);       // round-to-nearest-even
  return (unsigned short)(u >> 16);
}

__device__ __forceinline__ v8f wmma_bf16(v16bf a, v16bf b, v8f c) {
  return __builtin_amdgcn_wmma_f32_16x16x32_bf16(false, a, false, b, (short)0, c, false, false);
}

// ---------- fragment gathers (wave32, contiguous-K order, A and B alike), vectorized ----------
// A 16x32 bf16 tile at `tile` (row stride ld): VGPR v holds K pair kk,kk+1
// with kk = 2v + 16*half.
__device__ __forceinline__ v16bf frag_a16(const unsigned short* tile, int ld) {
  int lane = threadIdx.x & 31;
  int half = lane >> 4;
  const unsigned short* p = tile + (size_t)(lane & 15) * ld + 16 * half;
  v4u lo = *(const v4u*)(p);        // K = 16*half    .. +7
  v4u hi = *(const v4u*)(p + 8);    // K = 16*half + 8.. +7
  v8u u;
#pragma unroll
  for (int v = 0; v < 4; ++v) { u[v] = lo[v]; u[v + 4] = hi[v]; }
  return as_v16bf(u);
}

// B 32x16 bf16, source K-contiguous: element (k,n) = tile[n*ld + k].
// N = lane&15; VGPR v holds K = 2v + 16*half, +1.
__device__ __forceinline__ v16bf frag_b16(const unsigned short* tile, int ld) {
  int lane = threadIdx.x & 31;
  int half = lane >> 4;
  const unsigned short* p = tile + (size_t)(lane & 15) * ld + 16 * half;
  v4u lo = *(const v4u*)(p);        // K = 16*half    .. +7
  v4u hi = *(const v4u*)(p + 8);    // K = 16*half + 8.. +7
  v8u u;
#pragma unroll
  for (int v = 0; v < 4; ++v) { u[v] = lo[v]; u[v + 4] = hi[v]; }
  return as_v16bf(u);
}

// B 32x16 bf16, source N-contiguous (V tile in LDS [32][ld]):
// element (k,n) = tile[k*ld + n_off + n].
__device__ __forceinline__ v16bf frag_b_nmaj(const unsigned short* tile, int ld, int n_off) {
  int lane = threadIdx.x & 31;
  int half = lane >> 4;
  int n    = n_off + (lane & 15);
  v8u u;
#pragma unroll
  for (int v = 0; v < 8; ++v) {
    int kd = 2 * v + 16 * half;
    unsigned int lo = tile[kd * ld + n];
    unsigned int hi = tile[(kd + 1) * ld + n];
    u[v] = lo | (hi << 16);
  }
  return as_v16bf(u);
}

// ---------- fp32 -> bf16 conversion ----------
__global__ void cvt_f32_to_bf16(const float* __restrict__ in,
                                unsigned short* __restrict__ out, int n) {
  int i = blockIdx.x * blockDim.x + threadIdx.x;
  if (i < n) out[i] = f32_to_bf16_bits(in[i]);
}

// ---------- WMMA GEMM: C[M,N] = A[M,K](bf16) * B[N,K](bf16)^T ----------
// 256 threads = 8 waves. Workgroup tile 128x128; wave tile 32x64.
// K staged in 64-deep double-buffered LDS slabs via async b128 copies:
// 16 WMMAs per barrier pair.
template <bool OUT_BF16>
__global__ __launch_bounds__(256) void gemm_bf16_wmma(
    const unsigned short* __restrict__ A, const unsigned short* __restrict__ B,
    void* __restrict__ Cout, int M, int N, int K) {
  __shared__ unsigned short As[2][128 * 64];  // 16 KB each
  __shared__ unsigned short Bs[2][128 * 64];  // 16 KB each

  int tid  = threadIdx.x;
  int wave = tid >> 5;
  int wm   = wave & 3;        // 0..3 -> M offset
  int wn   = wave >> 2;       // 0..1 -> N offset
  int m0   = blockIdx.y * 128;
  int n0   = blockIdx.x * 128;

  // stage one 128x64 bf16 slab: 1024 x 16B chunks over 256 threads (4 each)
  auto stage = [&](const unsigned short* G, int ld, unsigned short* Ls, int k0) {
#pragma unroll
    for (int i = 0; i < 4; ++i) {
      int c = tid + 256 * i;             // 0..1023
      int row = c >> 3, cq = c & 7;      // 8 x 16B per 128B row
      copy16_g2l(G + (size_t)row * ld + k0 + cq * 8, Ls + row * 64 + cq * 8);
    }
  };

  v8f acc[2][4] = {};
  int buf = 0;
  stage(A + (size_t)m0 * K, K, As[0], 0);
  stage(B + (size_t)n0 * K, K, Bs[0], 0);

  int steps = K / 64;
  for (int s = 0; s < steps; ++s) {
    bool nxt = (s + 1) < steps;
    if (nxt) {            // prefetch next slab into other buffer (async)
      stage(A + (size_t)m0 * K, K, As[buf ^ 1], (s + 1) * 64);
      stage(B + (size_t)n0 * K, K, Bs[buf ^ 1], (s + 1) * 64);
      wait_async_le<8>(); // my 8 current-buffer chunks have landed
    } else {
      wait_async_le<0>();
    }
    __syncthreads();      // everyone's current-buffer chunks are in LDS

#pragma unroll
    for (int kk = 0; kk < 64; kk += 32) {
      v16bf a0 = frag_a16(&As[buf][(wm * 32) * 64 + kk], 64);
      v16bf a1 = frag_a16(&As[buf][(wm * 32 + 16) * 64 + kk], 64);
      v16bf b0 = frag_b16(&Bs[buf][(wn * 64) * 64 + kk], 64);
      v16bf b1 = frag_b16(&Bs[buf][(wn * 64 + 16) * 64 + kk], 64);
      v16bf b2 = frag_b16(&Bs[buf][(wn * 64 + 32) * 64 + kk], 64);
      v16bf b3 = frag_b16(&Bs[buf][(wn * 64 + 48) * 64 + kk], 64);
      acc[0][0] = wmma_bf16(a0, b0, acc[0][0]);
      acc[1][0] = wmma_bf16(a1, b0, acc[1][0]);
      acc[0][1] = wmma_bf16(a0, b1, acc[0][1]);
      acc[1][1] = wmma_bf16(a1, b1, acc[1][1]);
      acc[0][2] = wmma_bf16(a0, b2, acc[0][2]);
      acc[1][2] = wmma_bf16(a1, b2, acc[1][2]);
      acc[0][3] = wmma_bf16(a0, b3, acc[0][3]);
      acc[1][3] = wmma_bf16(a1, b3, acc[1][3]);
    }
    __syncthreads();      // all reads of buf done before it is re-staged
    buf ^= 1;
  }

  int lane = tid & 31, half = lane >> 4, nl = lane & 15;
#pragma unroll
  for (int ti = 0; ti < 2; ++ti)
#pragma unroll
    for (int tj = 0; tj < 4; ++tj)
#pragma unroll
      for (int r = 0; r < 8; ++r) {
        int m = m0 + wm * 32 + 16 * ti + r + 8 * half; // C layout: M = r + 8*half
        int n = n0 + wn * 64 + 16 * tj + nl;           //           N = lane & 15
        float val = acc[ti][tj][r];
        if (OUT_BF16)
          ((unsigned short*)Cout)[(size_t)m * N + n] = f32_to_bf16_bits(val);
        else
          ((float*)Cout)[(size_t)m * N + n] = val;
      }
}

// ---------- row reductions over the 16-lane half holding one C-row ----------
__device__ __forceinline__ float rowmax16(float v) {
#pragma unroll
  for (int off = 8; off >= 1; off >>= 1) v = fmaxf(v, __shfl_xor(v, off, 32));
  return v;
}
__device__ __forceinline__ float rowsum16(float v) {
#pragma unroll
  for (int off = 8; off >= 1; off >>= 1) v += __shfl_xor(v, off, 32);
  return v;
}

// ---------- Causal flash attention ----------
// qkv: [B*T, 3*D] bf16 (Q|K|V column blocks). Obf: [B*T, D] bf16.
// One wave per 16-row Q tile; 32 keys/step; V chunks double-buffered per wave
// in LDS via async copies; online softmax in fp32.
__global__ __launch_bounds__(128) void attn_wmma(
    const unsigned short* __restrict__ qkv, unsigned short* __restrict__ Obf) {
  const int T = 2048, D3 = 3072, DM = 1024, DH = 64;
  __shared__ unsigned short Pbuf[4][16 * 32];      // 4 KB: P relayout (per wave)
  __shared__ unsigned short Vs[4][2][32 * 64];     // 32 KB: V chunks (per wave, x2)

  int wave = threadIdx.x >> 5;
  int lane = threadIdx.x & 31;
  int half = lane >> 4, nl = lane & 15;
  int h = blockIdx.y, b = blockIdx.z;
  int q0 = blockIdx.x * 64 + wave * 16;

  const unsigned short* Q  = qkv + (size_t)(b * T) * D3 + h * DH;
  const unsigned short* Kp = Q + DM;
  const unsigned short* Vp = Q + 2 * DM;

  // stage one 32-key x 64-d V chunk: each lane copies its row (8 x 16B)
  auto stage_v = [&](unsigned short* dst, int kb) {
    const unsigned short* src = Vp + (size_t)(kb + lane) * D3;
    unsigned short* d = dst + lane * 64;
#pragma unroll
    for (int c = 0; c < 8; ++c) copy16_g2l(src + c * 8, d + c * 8);
  };

  // Q fragments (d = 0..31, 32..63), loaded once
  v16bf aQ0 = frag_a16(Q + (size_t)q0 * D3, D3);
  v16bf aQ1 = frag_a16(Q + (size_t)q0 * D3 + 32, D3);

  v8f   o[4] = {};
  float m_i[8], l_i[8];
#pragma unroll
  for (int r = 0; r < 8; ++r) { m_i[r] = -1e30f; l_i[r] = 0.f; }
  const float sc = 0.125f; // 1/sqrt(64)
  unsigned short* P = &Pbuf[wave][0];

  int cur = 0;
  stage_v(&Vs[wave][0][0], 0);

  for (int kb = 0; kb < q0 + 16; kb += 32) {
    bool nxt = (kb + 32) < (q0 + 16);
    if (nxt) { stage_v(&Vs[wave][cur ^ 1][0], kb + 32); wait_async_le<8>(); }
    else     { wait_async_le<0>(); }

    // Preload all four K fragments so their loads overlap the WMMA chain
    v16bf bk0 = frag_b16(Kp + (size_t)kb * D3, D3);
    v16bf bk1 = frag_b16(Kp + (size_t)kb * D3 + 32, D3);
    v16bf bk2 = frag_b16(Kp + (size_t)(kb + 16) * D3, D3);
    v16bf bk3 = frag_b16(Kp + (size_t)(kb + 16) * D3 + 32, D3);

    // S = Q * K^T  (two 16-key tiles, d=64 -> 2 WMMAs each)
    v8f s0 = {}, s1 = {};
    s0 = wmma_bf16(aQ0, bk0, s0);
    s1 = wmma_bf16(aQ0, bk2, s1);
    s0 = wmma_bf16(aQ1, bk1, s0);
    s1 = wmma_bf16(aQ1, bk3, s1);

    float alpha[8];
#pragma unroll
    for (int r = 0; r < 8; ++r) {
      int   mg = q0 + r + 8 * half;           // global query row of this slot
      int   n0 = kb + nl, n1 = kb + 16 + nl;  // global key columns
      float e0 = (n0 <= mg) ? s0[r] * sc : -1e30f;
      float e1 = (n1 <= mg) ? s1[r] * sc : -1e30f;
      float mx   = rowmax16(fmaxf(e0, e1));
      float mnew = fmaxf(m_i[r], mx);
      alpha[r]   = __expf(m_i[r] - mnew);
      float p0   = __expf(e0 - mnew);
      float p1   = __expf(e1 - mnew);
      l_i[r] = l_i[r] * alpha[r] + rowsum16(p0 + p1);
      m_i[r] = mnew;
      int row = r + 8 * half;                 // C layout -> LDS row-major [16][32]
      P[row * 32 + nl]      = f32_to_bf16_bits(p0);
      P[row * 32 + 16 + nl] = f32_to_bf16_bits(p1);
    }
#pragma unroll
    for (int t = 0; t < 4; ++t)
#pragma unroll
      for (int r = 0; r < 8; ++r) o[t][r] *= alpha[r];

    // O += P(16x32) * V(32x64): P re-read as A fragment; V from staged LDS
    v16bf aP = frag_a16(P, 32);
#pragma unroll
    for (int t = 0; t < 4; ++t) {
      v16bf bV = frag_b_nmaj(&Vs[wave][cur][0], 64, t * 16);
      o[t] = wmma_bf16(aP, bV, o[t]);
    }
    cur ^= 1;
  }

  // epilogue: normalize, store bf16 to O[B*T, D]
#pragma unroll
  for (int t = 0; t < 4; ++t)
#pragma unroll
    for (int r = 0; r < 8; ++r) {
      float val = o[t][r] / l_i[r];
      int row = b * T + q0 + r + 8 * half;
      int col = h * DH + t * 16 + nl;
      Obf[(size_t)row * DM + col] = f32_to_bf16_bits(val);
    }
}

// ---------- launch ----------
extern "C" void kernel_launch(void* const* d_in, const int* in_sizes, int n_in,
                              void* d_out, int out_size, void* d_ws, size_t ws_size,
                              hipStream_t stream) {
  const float* x     = (const float*)d_in[0]; // [2,2048,1024]
  const float* w_qkv = (const float*)d_in[1]; // [3072,1024]
  const float* w_out = (const float*)d_in[2]; // [1024,1024]

  const int B = 2, T = 2048, D = 1024, H = 16;
  const int MT = B * T; // 4096 rows

  char* ws = (char*)d_ws;
  unsigned short* xbf    = (unsigned short*)ws; ws += (size_t)MT * D * 2;     // 8 MB
  unsigned short* wqkvbf = (unsigned short*)ws; ws += (size_t)3 * D * D * 2;  // 6 MB
  unsigned short* woutbf = (unsigned short*)ws; ws += (size_t)D * D * 2;      // 2 MB
  unsigned short* qkvbf  = (unsigned short*)ws; ws += (size_t)MT * 3 * D * 2; // 24 MB
  unsigned short* obf    = (unsigned short*)ws;                               // 8 MB

  int n1 = MT * D, n2 = 3 * D * D, n3 = D * D;
  cvt_f32_to_bf16<<<(n1 + 255) / 256, 256, 0, stream>>>(x, xbf, n1);
  cvt_f32_to_bf16<<<(n2 + 255) / 256, 256, 0, stream>>>(w_qkv, wqkvbf, n2);
  cvt_f32_to_bf16<<<(n3 + 255) / 256, 256, 0, stream>>>(w_out, woutbf, n3);

  // QKV projection: [4096,3072] = x * w_qkv^T
  gemm_bf16_wmma<true><<<dim3(3 * D / 128, MT / 128), 256, 0, stream>>>(
      xbf, wqkvbf, qkvbf, MT, 3 * D, D);

  // Causal MHSA
  attn_wmma<<<dim3(T / 64, H, B), 128, 0, stream>>>(qkvbf, obf);

  // Output projection: d_out (fp32) = O * w_out^T
  gemm_bf16_wmma<false><<<dim3(D / 128, MT / 128), 256, 0, stream>>>(
      obf, woutbf, d_out, MT, D, D);
}
